// Tiresias_68616397521079
// MI455X (gfx1250) — matrix operand order repaired and checked by reading, not compiled, not run on hardware
//
#include <hip/hip_runtime.h>
#include <hip/hip_bf16.h>

// ---- problem constants ----
#define HIDDEN   128
#define KCELL    4
#define GATES    2048        // 4*HIDDEN*KCELL
#define BATCH    256
#define SEQ      256
#define VOCAB    1024
#define OUT_N    1024
#define BT       16          // batch tile per block

#define LOG2E 1.4426950408889634f
#define LN2   0.6931471805599453f

typedef __attribute__((ext_vector_type(16))) __bf16 v16bf;
typedef __attribute__((ext_vector_type(8)))  float  v8f;

union FragBF {
    v16bf v;
    unsigned short u[16];
    unsigned int   d[8];
    uint4          q[2];
};

__device__ __forceinline__ unsigned short f2bf(float f) {
    union { float f; unsigned u; } x; x.f = f;
    unsigned r = x.u + 0x7FFFu + ((x.u >> 16) & 1u);   // round-to-nearest-even
    return (unsigned short)(r >> 16);
}

// fast transcendental helpers: single v_exp_f32 / v_rcp_f32 / v_log_f32, no IEEE div
__device__ __forceinline__ float fexp(float x)  { return __builtin_amdgcn_exp2f(x * LOG2E); }
__device__ __forceinline__ float frcp(float x)  { return __builtin_amdgcn_rcpf(x); }
__device__ __forceinline__ float fsigmoid(float x) { return frcp(1.0f + fexp(-x)); }
__device__ __forceinline__ float ftanh(float x) { return 2.0f * frcp(1.0f + fexp(-2.0f * x)) - 1.0f; }
__device__ __forceinline__ float flog(float x)  { return __builtin_amdgcn_logf(x) * LN2; }

// A fragment: 16x32 bf16 tile of h (row = batch), from LDS bf16 buffer hs[16][HIDDEN].
// Layout: lanes 0-15 hold row m with K = 0..15; lanes 16-31 hold row m
// with K = 16..31; VGPR j holds K pair 2j,2j+1 (matches load_B_frag).
__device__ __forceinline__ v16bf load_A_frag(const unsigned short* hs, int lane, int kb) {
    const int m   = lane & 15;
    const int hi8 = ((lane >> 4) & 1) * 16;
    const unsigned int* hsu = (const unsigned int*)hs;   // 2 bf16 per dword
    FragBF f;
#pragma unroll
    for (int j = 0; j < 8; ++j) {
        int k = kb * 32 + hi8 + (j << 1);
        f.d[j] = hsu[(m * HIDDEN + k) >> 1];
    }
    return f.v;
}

// B fragment: 32x16 bf16 tile of W^T, i.e. B[k][n] = W[g0+n][k] with W row-major (rows=out,
// cols=HIDDEN, bf16). Lane n holds column n: lanes 0-15 K=kb*32+0..15, lanes 16-31 K=+16..31.
// 16 contiguous bf16 per lane -> two global_load_b128.
__device__ __forceinline__ v16bf load_B_frag(const unsigned short* W, int g0, int lane, int kb) {
    const int n     = lane & 15;
    const int kbase = kb * 32 + ((lane >> 4) & 1) * 16;
    const uint4* p = (const uint4*)(W + (size_t)(g0 + n) * HIDDEN + kbase);
    FragBF f;
    f.q[0] = p[0];
    f.q[1] = p[1];
    return f.v;
}

// ---------------------------------------------------------------------------
// Prologue: quantize W_hh / W_out to bf16 in workspace; zero h and c state.
// ---------------------------------------------------------------------------
__global__ void prep_kernel(const float* __restrict__ W_hh, const float* __restrict__ W_out,
                            unsigned short* __restrict__ Whh_bf, unsigned short* __restrict__ Wout_bf,
                            float* __restrict__ h, float* __restrict__ c) {
    const size_t i = (size_t)blockIdx.x * blockDim.x + threadIdx.x;
    if (i < (size_t)GATES * HIDDEN)  Whh_bf[i]  = f2bf(W_hh[i]);
    if (i < (size_t)OUT_N * HIDDEN)  Wout_bf[i] = f2bf(W_out[i]);
    if (i < (size_t)BATCH * HIDDEN)  h[i] = 0.0f;
    if (i < (size_t)BATCH * KCELL * HIDDEN) c[i] = 0.0f;
}

// ---------------------------------------------------------------------------
// One LSTM timestep. Grid: BATCH/16 blocks x 512 threads (16 waves).
// Each block owns 16 batch rows and all 2048 gate rows.
// ---------------------------------------------------------------------------
__global__ __launch_bounds__(512) void lstm_step_kernel(
        const int*   __restrict__ X,
        const float* __restrict__ W_ih,
        const float* __restrict__ b_ih,
        const float* __restrict__ b_hh,
        const unsigned short* __restrict__ Whh_bf,
        float* __restrict__ h_state,
        float* __restrict__ c_state,
        int t) {
    extern __shared__ char smem[];
    float*          gates = (float*)smem;                                   // 16*2048 f32 (128 KB)
    unsigned short* hs    = (unsigned short*)(smem + BT * GATES * 4);       // 16*128 bf16 (4 KB)
    int*            xidx  = (int*)(smem + BT * GATES * 4 + BT * HIDDEN * 2);// 16
    float*          osum  = (float*)(xidx + BT);                            // 16*4
    int*            ksel  = (int*)(osum + BT * KCELL);                      // 16

    const int tid = threadIdx.x;
    const int b0  = blockIdx.x * BT;

    // phase 0: h -> bf16 LDS, event indices, clear o-sums
    for (int i = tid; i < BT * HIDDEN; i += blockDim.x)
        hs[i] = f2bf(h_state[(size_t)b0 * HIDDEN + i]);
    if (tid < BT) xidx[tid] = X[(size_t)(b0 + tid) * SEQ + t];
    if (tid < BT * KCELL) osum[tid] = 0.0f;
    __syncthreads();

    // phase 1: gates = xp + h @ W_hh^T + b  via bf16 WMMA, then activations -> LDS
    const int wave = tid >> 5;
    const int lane = tid & 31;
    const int n    = lane & 15;
    const int mhi  = ((lane >> 4) & 1) * 8;
    for (int ti = 0; ti < 8; ++ti) {                       // 16 waves * 8 tiles * 16 = 2048 gates
        const int g0 = (wave * 8 + ti) * 16;
        const int g  = g0 + n;                             // this lane's gate column
        const float bias = b_ih[g] + b_hh[g];
        // seed C with the one-hot input projection gather + biases
        v8f acc;
#pragma unroll
        for (int r = 0; r < 8; ++r)
            acc[r] = W_ih[(size_t)g * VOCAB + xidx[r + mhi]] + bias;
#pragma unroll
        for (int kb = 0; kb < 4; ++kb) {
            v16bf a = load_A_frag(hs, lane, kb);
            v16bf b = load_B_frag(Whh_bf, g0, lane, kb);
            acc = __builtin_amdgcn_wmma_f32_16x16x32_bf16(
                      false, a, false, b, (short)0, acc, false, false);
        }
        // unified branchless activation: s=1 -> sigmoid(v); s=2 -> 2*sig(2v)-1 = tanh(v)
        const int   gtype = (g >> 7) & 3;                  // 0:i 1:f 2:g 3:o
        const float s     = (gtype == 2) ? 2.0f : 1.0f;
        const float sm1   = s - 1.0f;
#pragma unroll
        for (int r = 0; r < 8; ++r) {
            float v   = acc[r];
            float act = s * frcp(1.0f + fexp(-s * v)) - sm1;
            gates[(r + mhi) * GATES + g] = act;
        }
    }
    __syncthreads();

    // phase 2: c_new = f*c + i*g for all K cells; accumulate per-(b,k) o sums in LDS
    for (int i = tid; i < BT * HIDDEN; i += blockDim.x) {
        const int m  = i >> 7;            // local batch
        const int hh = i & (HIDDEN - 1);
        const int b  = b0 + m;
#pragma unroll
        for (int k = 0; k < KCELL; ++k) {
            const int base = m * GATES + k * (4 * HIDDEN);
            float iv = gates[base + 0 * HIDDEN + hh];
            float fv = gates[base + 1 * HIDDEN + hh];
            float gv = gates[base + 2 * HIDDEN + hh];
            float ov = gates[base + 3 * HIDDEN + hh];
            const size_t ci = ((size_t)b * KCELL + k) * HIDDEN + hh;
            float cn = fv * c_state[ci] + iv * gv;
            c_state[ci] = cn;                              // persist for next step
            gates[base + 2 * HIDDEN + hh] = cn;            // reuse dead g-slot as c_new cache
            atomicAdd(&osum[m * KCELL + k], ov);           // ds_add_f32
        }
    }
    __syncthreads();

    // phase 3: stochastic pooling index ~ softmax(osum) per batch (hash-based uniform)
    if (tid < BT) {
        float mx = osum[tid * KCELL];
#pragma unroll
        for (int k = 1; k < KCELL; ++k) mx = fmaxf(mx, osum[tid * KCELL + k]);
        float p[KCELL], z = 0.0f;
#pragma unroll
        for (int k = 0; k < KCELL; ++k) { p[k] = fexp(osum[tid * KCELL + k] - mx); z += p[k]; }
        unsigned s = (unsigned)(t * 2654435761u) ^ (unsigned)((b0 + tid) * 2246822519u) ^ 0x9E3779B9u;
        s ^= s >> 15; s *= 0x2C1B3C6Du; s ^= s >> 12; s *= 0x297A2D39u; s ^= s >> 15;
        float u = (float)(s >> 8) * (1.0f / 16777216.0f) * z;
        int k = 0; float cum = p[0];
        while (k < KCELL - 1 && u > cum) { ++k; cum += p[k]; }
        ksel[tid] = k;
    }
    __syncthreads();

    // phase 4: h = o_sel * tanh(c_sel)
    for (int i = tid; i < BT * HIDDEN; i += blockDim.x) {
        const int m  = i >> 7;
        const int hh = i & (HIDDEN - 1);
        const int k  = ksel[m];
        const int base = m * GATES + k * (4 * HIDDEN);
        float ov = gates[base + 3 * HIDDEN + hh];
        float cn = gates[base + 2 * HIDDEN + hh];
        h_state[(size_t)(b0 + m) * HIDDEN + hh] = ov * ftanh(cn);
    }
}

// ---------------------------------------------------------------------------
// logits = h @ W_out^T + b_out ; log_softmax over 1024. Grid: BATCH/16 x 256.
// ---------------------------------------------------------------------------
__global__ __launch_bounds__(256) void output_kernel(
        const float* __restrict__ h_state,
        const unsigned short* __restrict__ Wout_bf,
        const float* __restrict__ b_out,
        float* __restrict__ out) {
    extern __shared__ char smem[];
    float*          logit  = (float*)smem;                                  // 16*1024 (64 KB)
    unsigned short* hs     = (unsigned short*)(smem + BT * OUT_N * 4);      // 16*128 bf16
    float*          red    = (float*)(smem + BT * OUT_N * 4 + BT * HIDDEN * 2); // 256
    float*          rowmax = red + 256;                                     // 16
    float*          rowlse = rowmax + BT;                                   // 16

    const int tid = threadIdx.x;
    const int b0  = blockIdx.x * BT;

    for (int i = tid; i < BT * HIDDEN; i += blockDim.x)
        hs[i] = f2bf(h_state[(size_t)b0 * HIDDEN + i]);
    __syncthreads();

    const int wave = tid >> 5;
    const int lane = tid & 31;
    const int n    = lane & 15;
    const int mhi  = ((lane >> 4) & 1) * 8;
    for (int ti = 0; ti < 8; ++ti) {                       // 8 waves * 8 tiles * 16 = 1024 outputs
        const int g0 = (wave * 8 + ti) * 16;
        v8f acc = {};
#pragma unroll
        for (int kb = 0; kb < 4; ++kb) {
            v16bf a = load_A_frag(hs, lane, kb);
            v16bf b = load_B_frag(Wout_bf, g0, lane, kb);
            acc = __builtin_amdgcn_wmma_f32_16x16x32_bf16(
                      false, a, false, b, (short)0, acc, false, false);
        }
        const int g = g0 + n;
        const float bias = b_out[g];
#pragma unroll
        for (int r = 0; r < 8; ++r)
            logit[(r + mhi) * OUT_N + g] = acc[r] + bias;
    }
    __syncthreads();

    // log-softmax: 16 threads cooperate per batch row
    const int row = tid >> 4;
    const int c16 = tid & 15;
    float mx = -3.4e38f;
    for (int j = c16; j < OUT_N; j += 16) mx = fmaxf(mx, logit[row * OUT_N + j]);
    red[tid] = mx;
    __syncthreads();
    if (c16 == 0) {
        float m2 = red[row * 16];
        for (int j = 1; j < 16; ++j) m2 = fmaxf(m2, red[row * 16 + j]);
        rowmax[row] = m2;
    }
    __syncthreads();
    const float rm = rowmax[row];
    float sm = 0.0f;
    for (int j = c16; j < OUT_N; j += 16) sm += fexp(logit[row * OUT_N + j] - rm);
    red[tid] = sm;
    __syncthreads();
    if (c16 == 0) {
        float s2 = 0.0f;
        for (int j = 0; j < 16; ++j) s2 += red[row * 16 + j];
        rowlse[row] = flog(s2);
    }
    __syncthreads();
    const float lz = rowlse[row] + rm;
    for (int j = c16; j < OUT_N; j += 16)
        out[(size_t)(b0 + row) * OUT_N + j] = logit[row * OUT_N + j] - lz;
}

// ---------------------------------------------------------------------------
extern "C" void kernel_launch(void* const* d_in, const int* in_sizes, int n_in,
                              void* d_out, int out_size, void* d_ws, size_t ws_size,
                              hipStream_t stream) {
    (void)in_sizes; (void)n_in; (void)out_size; (void)ws_size;
    const int*   X     = (const int*)  d_in[0];
    const float* W_ih  = (const float*)d_in[1];
    const float* b_ih  = (const float*)d_in[2];
    const float* W_hh  = (const float*)d_in[3];
    const float* b_hh  = (const float*)d_in[4];
    const float* W_out = (const float*)d_in[5];
    const float* b_out = (const float*)d_in[6];
    float* out = (float*)d_out;

    char* ws = (char*)d_ws;
    float* h = (float*)ws;                 ws += (size_t)BATCH * HIDDEN * 4;
    float* c = (float*)ws;                 ws += (size_t)BATCH * KCELL * HIDDEN * 4;
    unsigned short* Whh_bf  = (unsigned short*)ws;  ws += (size_t)GATES * HIDDEN * 2;
    unsigned short* Wout_bf = (unsigned short*)ws;  ws += (size_t)OUT_N * HIDDEN * 2;

    const size_t step_lds = (size_t)BT * GATES * 4 + (size_t)BT * HIDDEN * 2
                          + BT * 4 + BT * KCELL * 4 + BT * 4;
    const size_t out_lds  = (size_t)BT * OUT_N * 4 + (size_t)BT * HIDDEN * 2
                          + (256 + BT + BT) * 4;

    static bool attr_set = false;   // attribute config, not work: deterministic
    if (!attr_set) {
        hipFuncSetAttribute((const void*)lstm_step_kernel,
                            hipFuncAttributeMaxDynamicSharedMemorySize, (int)step_lds);
        hipFuncSetAttribute((const void*)output_kernel,
                            hipFuncAttributeMaxDynamicSharedMemorySize, (int)out_lds);
        attr_set = true;
    }

    prep_kernel<<<(GATES * HIDDEN + 255) / 256, 256, 0, stream>>>(
        W_hh, W_out, Whh_bf, Wout_bf, h, c);

    for (int t = 0; t < SEQ; ++t) {
        lstm_step_kernel<<<BATCH / BT, 512, step_lds, stream>>>(
            X, W_ih, b_ih, b_hh, Whh_bf, h, c, t);
    }

    output_kernel<<<BATCH / BT, 256, out_lds, stream>>>(h, Wout_bf, b_out, out);
}
